// MoTAttention_54700703482359
// MI455X (gfx1250) — compile-verified, matrix operand order checked by reading, not run on hardware
//
#include <hip/hip_runtime.h>

// ---------------- problem constants ----------------
#define BB   4
#define SS   1024
#define DD   1024
#define HH   16
#define HD   64
#define EE   4
#define NN   (BB * SS)      // 4096
#define F3   (3 * DD)       // 3072

typedef __attribute__((ext_vector_type(16))) __bf16        v16bf;
typedef __attribute__((ext_vector_type(8)))  float         v8f;
typedef __attribute__((ext_vector_type(4)))  unsigned int  v4u;
typedef __attribute__((ext_vector_type(8)))  int           v8i;
typedef __attribute__((ext_vector_type(4)))  int           v4i;

#if defined(__gfx1250__) && __has_builtin(__builtin_amdgcn_tensor_load_to_lds)
#define HAVE_TDM 1
#else
#define HAVE_TDM 0
#endif

__device__ __forceinline__ void wave_lds_fence() {
    asm volatile("s_wait_dscnt 0" ::: "memory");
    __builtin_amdgcn_wave_barrier();
}

#if defined(__gfx1250__)
__device__ __forceinline__ unsigned lds_offset_of(const void* p) {
    return (unsigned)(size_t)(__attribute__((address_space(3))) const void*)p;
}
__device__ __forceinline__ void wait_tensorcnt0() {
#if __has_builtin(__builtin_amdgcn_s_wait_tensorcnt)
    __builtin_amdgcn_s_wait_tensorcnt(0);
#else
    asm volatile("s_wait_tensorcnt 0x0" ::: "memory");
#endif
}
#endif

// ---------------- kernel 0: fp32 -> bf16 weight convert + transpose ----------------
// in:  [E][K][F] fp32 (k-major rows)   out: [E][F][K] bf16 (f-major rows)
__global__ void convert_transpose_kernel(const float* __restrict__ in,
                                         __bf16* __restrict__ out,
                                         int K, int F) {
    size_t i = (size_t)blockIdx.x * 256 + threadIdx.x;
    if (i >= (size_t)EE * K * F) return;
    int    f = (int)(i % F);
    size_t t = i / F;
    int    k = (int)(t % K);
    int    e = (int)(t / K);
    out[((size_t)e * F + f) * K + k] = (__bf16)in[i];
}

// ---------------- kernel 1: router softmax + layernorm ----------------
__global__ __launch_bounds__(256) void router_ln_kernel(
    const float* __restrict__ x, const float* __restrict__ Wr,
    const float* __restrict__ br, float* __restrict__ score,
    float* __restrict__ xn) {
    const int n   = blockIdx.x;
    const int tid = threadIdx.x;
    const float* xp = x + (size_t)n * DD;

    float4 v = ((const float4*)xp)[tid];
    float va[4] = {v.x, v.y, v.z, v.w};

    float sum = 0.f, sq = 0.f, dot[EE] = {0.f, 0.f, 0.f, 0.f};
    const int d0 = tid * 4;
    #pragma unroll
    for (int j = 0; j < 4; ++j) {
        float xv = va[j];
        sum += xv;
        sq  += xv * xv;
        const float* wr = Wr + (size_t)(d0 + j) * EE;
        #pragma unroll
        for (int e = 0; e < EE; ++e) dot[e] += xv * wr[e];
    }

    __shared__ float red[6][256];
    red[0][tid] = sum; red[1][tid] = sq;
    #pragma unroll
    for (int e = 0; e < EE; ++e) red[2 + e][tid] = dot[e];
    __syncthreads();
    for (int s = 128; s > 0; s >>= 1) {
        if (tid < s) {
            #pragma unroll
            for (int i = 0; i < 6; ++i) red[i][tid] += red[i][tid + s];
        }
        __syncthreads();
    }

    __shared__ float stats[2];
    if (tid == 0) {
        float mu  = red[0][0] * (1.f / DD);
        float var = red[1][0] * (1.f / DD) - mu * mu;
        stats[0] = mu;
        stats[1] = rsqrtf(var + 1e-5f);
        float lg[EE], mx = -1e30f;
        #pragma unroll
        for (int e = 0; e < EE; ++e) { lg[e] = red[2 + e][0] + br[e]; mx = fmaxf(mx, lg[e]); }
        float den = 0.f;
        #pragma unroll
        for (int e = 0; e < EE; ++e) { lg[e] = __expf(lg[e] - mx); den += lg[e]; }
        float inv = 1.f / den;
        #pragma unroll
        for (int e = 0; e < EE; ++e) score[(size_t)n * EE + e] = lg[e] * inv;
    }
    __syncthreads();

    const float mu = stats[0], rsig = stats[1];
    float4 o;
    o.x = (va[0] - mu) * rsig; o.y = (va[1] - mu) * rsig;
    o.z = (va[2] - mu) * rsig; o.w = (va[3] - mu) * rsig;
    ((float4*)(xn + (size_t)n * DD))[tid] = o;
}

// ---------------- kernel 2/4: score-folded multi-expert GEMM ----------------
// C[N,F] = sum_e (score[:,e] * (A*g_e + b_e)) @ W_e  + sum_e score*bias_e
// W is PRE-TRANSPOSED bf16 [E][F][Kdim].  Tile 128x128, K-step 32.
// W tile is DMA'd by the Tensor Data Mover into LDS with hardware padding
// (pitch 36 bf16 = 16 DWORDs data + 2 DWORDs pad per f-row).
template <bool HAS_AFFINE>
__global__ __launch_bounds__(256) void moe_gemm_kernel(
    const float* __restrict__ A, const float* __restrict__ score,
    const float* __restrict__ gvec, const float* __restrict__ bvec,
    const __bf16* __restrict__ W, const float* __restrict__ bias,
    float* __restrict__ C, int Kdim, int F) {
    const int n0   = blockIdx.x * 128;
    const int f0   = blockIdx.y * 128;
    const int tid  = threadIdx.x;
    const int lane = tid & 31;
    const int wave = tid >> 5;

    __shared__ __bf16 As[128][36];   // [row][k]   (pitch 36 = 72B)
    __shared__ __bf16 Ws[128][36];   // [fcol][k]  (pitch 36 = 72B)

    const v8f vz = {0.f, 0.f, 0.f, 0.f, 0.f, 0.f, 0.f, 0.f};
    v8f acc[8];
    #pragma unroll
    for (int t = 0; t < 8; ++t) acc[t] = vz;

    const int m0  = wave * 16;
    const int kb8 = (lane & 16) ? 8 : 0;

    for (int e = 0; e < EE; ++e) {
        for (int k0 = 0; k0 < Kdim; k0 += 32) {
#if HAVE_TDM
            // ---- W tile via Tensor Data Mover: [128 f][32 k] bf16 ----
            if (tid == 0) {
                const unsigned long long ga = (unsigned long long)(size_t)(
                    W + ((size_t)e * F + f0) * Kdim + k0);
                v4u g0;
                g0.x = 1u;                                   // count=1, load desc
                g0.y = lds_offset_of(&Ws[0][0]);             // lds_addr
                g0.z = (unsigned)(ga & 0xFFFFFFFFull);       // global_addr lo
                g0.w = (unsigned)((ga >> 32) & 0x1FFFFFFull) // global_addr hi
                       | (2u << 30);                         // type = image
                v8i g1;
                g1[0] = (1 << 16)        // data_size = 2 bytes
                      | (1 << 20)        // pad_enable
                      | (3 << 22)        // pad_interval: every 16 DWORDs
                      | (1 << 25);       // pad_amount: 2 DWORDs
                g1[1] = (Kdim & 0xFFFF) << 16;                       // tensor_dim0 lo
                g1[2] = ((Kdim >> 16) & 0xFFFF) | ((F & 0xFFFF) << 16); // d0 hi | d1 lo
                g1[3] = ((F >> 16) & 0xFFFF) | (32 << 16);           // d1 hi | tile_dim0
                g1[4] = 128;                                          // tile_dim1 (dim2=0)
                g1[5] = Kdim;                                         // dim0_stride lo
                g1[6] = 0;
                g1[7] = 0;
                const v4i z4 = {0, 0, 0, 0};
                const v8i z8 = {0, 0, 0, 0, 0, 0, 0, 0};
                // 6-arg form (clang-23 / therock-10.0 headers): groups 2,3 + spare group
                __builtin_amdgcn_tensor_load_to_lds(g0, g1, z4, z4, z8, 0);
            }
#endif
            // ---- A tile: per-expert affine + score fold, fp32 -> bf16 ----
            {
                const int r  = tid >> 1;          // 0..127
                const int ks = (tid & 1) * 16;    // 0 | 16
                const int n  = n0 + r;
                const float s = score[(size_t)n * EE + e];
                const float4* ap = (const float4*)(A + (size_t)n * Kdim + k0 + ks);
                if (HAS_AFFINE) {
                    const float4* gp = (const float4*)(gvec + (size_t)e * Kdim + k0 + ks);
                    const float4* bp = (const float4*)(bvec + (size_t)e * Kdim + k0 + ks);
                    #pragma unroll
                    for (int q = 0; q < 4; ++q) {
                        const float4 av = ap[q], gv = gp[q], bv = bp[q];
                        As[r][ks + q * 4 + 0] = (__bf16)(s * (av.x * gv.x + bv.x));
                        As[r][ks + q * 4 + 1] = (__bf16)(s * (av.y * gv.y + bv.y));
                        As[r][ks + q * 4 + 2] = (__bf16)(s * (av.z * gv.z + bv.z));
                        As[r][ks + q * 4 + 3] = (__bf16)(s * (av.w * gv.w + bv.w));
                    }
                } else {
                    #pragma unroll
                    for (int q = 0; q < 4; ++q) {
                        const float4 av = ap[q];
                        As[r][ks + q * 4 + 0] = (__bf16)(s * av.x);
                        As[r][ks + q * 4 + 1] = (__bf16)(s * av.y);
                        As[r][ks + q * 4 + 2] = (__bf16)(s * av.z);
                        As[r][ks + q * 4 + 3] = (__bf16)(s * av.w);
                    }
                }
            }
#if HAVE_TDM
            if (tid == 0) wait_tensorcnt0();
#else
            // ---- fallback: manual W tile copy (W is [E][F][K], contiguous k) ----
            {
                const int fr = tid >> 1;          // 0..127
                const int kh = (tid & 1) * 16;    // 0 | 16
                const __bf16* wp = W + ((size_t)e * F + f0 + fr) * Kdim + k0 + kh;
                #pragma unroll
                for (int j = 0; j < 16; ++j) Ws[fr][kh + j] = wp[j];
                if (k0 + 32 < Kdim) __builtin_prefetch(wp + 32, 0, 0);
            }
#endif
            __syncthreads();

            // ---- fragments + 8 WMMAs ----
            v16bf afrag;
            const int arow = m0 + (lane & 15);
            #pragma unroll
            for (int j = 0; j < 16; ++j) {
                const int kk = ((j >> 3) * 16) + kb8 + (j & 7);
                afrag[j] = As[arow][kk];
            }
            #pragma unroll
            for (int t = 0; t < 8; ++t) {
                v16bf bfrag;
                const int fl = t * 16 + (lane & 15);
                #pragma unroll
                for (int j = 0; j < 16; ++j) {
                    const int kk = ((j >> 3) * 16) + kb8 + (j & 7);
                    bfrag[j] = Ws[fl][kk];
                }
                acc[t] = __builtin_amdgcn_wmma_f32_16x16x32_bf16(
                    false, afrag, false, bfrag, (short)0, acc[t], false, false);
            }
            __syncthreads();
        }
    }

    // ---- epilogue: + sum_e score*bias_e, store fp32 ----
    #pragma unroll
    for (int r = 0; r < 8; ++r) {
        const int row_local = m0 + r + kb8;
        const int n = n0 + row_local;
        float sc[EE];
        #pragma unroll
        for (int e = 0; e < EE; ++e) sc[e] = score[(size_t)n * EE + e];
        #pragma unroll
        for (int t = 0; t < 8; ++t) {
            const int f = f0 + t * 16 + (lane & 15);
            float v = acc[t][r];
            #pragma unroll
            for (int e = 0; e < EE; ++e) v += sc[e] * bias[(size_t)e * F + f];
            C[(size_t)n * F + f] = v;
        }
    }
}

// ---------------- kernel 3: causal flash attention, WMMA bf16 ----------------
__global__ __launch_bounds__(128) void attn_kernel(
    const float* __restrict__ qkv, float* __restrict__ ctx) {
    const int bh = blockIdx.x;
    const int b  = bh >> 4;
    const int h  = bh & 15;
    const int q0 = blockIdx.y * 64;
    const int tid  = threadIdx.x;
    const int lane = tid & 31;
    const int wave = tid >> 5;

    __shared__ __bf16 Qs[64][64];
    __shared__ __bf16 Ks[4][32][64];
    __shared__ __bf16 Vs[4][32][64];
    __shared__ __bf16 Ps[4][16][32];

    for (int i = tid; i < 64 * 64; i += 128) {
        const int r = i >> 6, c = i & 63;
        const float qv = qkv[(size_t)(b * SS + q0 + r) * F3 + h * HD + c];
        Qs[r][c] = (__bf16)(qv * 0.125f);
    }
    __syncthreads();

    const int qw  = q0 + wave * 16;
    const int kb8 = (lane & 16) ? 8 : 0;

    const v8f vz = {0.f, 0.f, 0.f, 0.f, 0.f, 0.f, 0.f, 0.f};
    v8f o[4];
    #pragma unroll
    for (int t = 0; t < 4; ++t) o[t] = vz;
    float mrow[8], lrow[8];
    #pragma unroll
    for (int r = 0; r < 8; ++r) { mrow[r] = -1e30f; lrow[r] = 0.f; }

    const int nchunk = (qw + 15) / 32 + 1;

    for (int c = 0; c < nchunk; ++c) {
        const int kstart = c * 32;
        for (int i = lane; i < 32 * 64; i += 32) {
            const int kk = i >> 6, hd = i & 63;
            const size_t base = (size_t)(b * SS + kstart + kk) * F3 + h * HD + hd;
            Ks[wave][kk][hd] = (__bf16)qkv[base + DD];
            Vs[wave][kk][hd] = (__bf16)qkv[base + 2 * DD];
        }
        wave_lds_fence();

        v8f sacc[2]; sacc[0] = vz; sacc[1] = vz;
        #pragma unroll
        for (int ks = 0; ks < 64; ks += 32) {
            v16bf aq;
            const int qr = wave * 16 + (lane & 15);
            #pragma unroll
            for (int j = 0; j < 16; ++j) {
                const int kk = ((j >> 3) * 16) + kb8 + (j & 7);
                aq[j] = Qs[qr][ks + kk];
            }
            #pragma unroll
            for (int sub = 0; sub < 2; ++sub) {
                v16bf bk;
                const int key = sub * 16 + (lane & 15);
                #pragma unroll
                for (int j = 0; j < 16; ++j) {
                    const int kk = ((j >> 3) * 16) + kb8 + (j & 7);
                    bk[j] = Ks[wave][key][ks + kk];
                }
                sacc[sub] = __builtin_amdgcn_wmma_f32_16x16x32_bf16(
                    false, aq, false, bk, (short)0, sacc[sub], false, false);
            }
        }

        #pragma unroll
        for (int r = 0; r < 8; ++r) {
            const int rowg = qw + r + kb8;
            float s0 = sacc[0][r], s1 = sacc[1][r];
            const int key0 = kstart + (lane & 15);
            if (key0 > rowg)      s0 = -1e30f;
            if (key0 + 16 > rowg) s1 = -1e30f;
            float mx = fmaxf(s0, s1);
            #pragma unroll
            for (int off = 8; off >= 1; off >>= 1)
                mx = fmaxf(mx, __shfl_xor(mx, off, 16));
            const float mnew = fmaxf(mrow[r], mx);
            const float p0 = __expf(s0 - mnew);
            const float p1 = __expf(s1 - mnew);
            float psum = p0 + p1;
            #pragma unroll
            for (int off = 8; off >= 1; off >>= 1)
                psum += __shfl_xor(psum, off, 16);
            const float alpha = __expf(mrow[r] - mnew);
            lrow[r] = lrow[r] * alpha + psum;
            mrow[r] = mnew;
            #pragma unroll
            for (int t = 0; t < 4; ++t) o[t][r] *= alpha;
            const int rl = r + kb8;
            Ps[wave][rl][lane & 15]        = (__bf16)p0;
            Ps[wave][rl][16 + (lane & 15)] = (__bf16)p1;
        }
        wave_lds_fence();

        v16bf ap;
        #pragma unroll
        for (int j = 0; j < 16; ++j) {
            const int kk = ((j >> 3) * 16) + kb8 + (j & 7);
            ap[j] = Ps[wave][lane & 15][kk];
        }
        #pragma unroll
        for (int t = 0; t < 4; ++t) {
            v16bf bv;
            const int hd = t * 16 + (lane & 15);
            #pragma unroll
            for (int j = 0; j < 16; ++j) {
                const int kk = ((j >> 3) * 16) + kb8 + (j & 7);
                bv[j] = Vs[wave][kk][hd];
            }
            o[t] = __builtin_amdgcn_wmma_f32_16x16x32_bf16(
                false, ap, false, bv, (short)0, o[t], false, false);
        }
    }

    #pragma unroll
    for (int r = 0; r < 8; ++r) {
        const int rowg = qw + r + kb8;
        const float inv = 1.f / lrow[r];
        #pragma unroll
        for (int t = 0; t < 4; ++t)
            ctx[(size_t)(b * SS + rowg) * DD + h * HD + t * 16 + (lane & 15)] =
                o[t][r] * inv;
    }
}

// ---------------- host launcher ----------------
extern "C" void kernel_launch(void* const* d_in, const int* in_sizes, int n_in,
                              void* d_out, int out_size, void* d_ws, size_t ws_size,
                              hipStream_t stream) {
    const float* x      = (const float*)d_in[0];
    const float* Wr     = (const float*)d_in[1];
    const float* br     = (const float*)d_in[2];
    const float* ln_g   = (const float*)d_in[3];
    const float* ln_b   = (const float*)d_in[4];
    const float* W_attn = (const float*)d_in[5];
    const float* b_attn = (const float*)d_in[6];
    const float* W_proj = (const float*)d_in[7];
    const float* b_proj = (const float*)d_in[8];
    float* out = (float*)d_out;

    char* ws = (char*)d_ws;
    size_t off = 0;
    float*  score  = (float*)(ws + off);  off += (size_t)NN * EE * 4;
    float*  xn     = (float*)(ws + off);  off += (size_t)NN * DD * 4;
    float*  qkvb   = (float*)(ws + off);  off += (size_t)NN * F3 * 4;
    float*  ctx    = (float*)(ws + off);  off += (size_t)NN * DD * 4;
    __bf16* WattnT = (__bf16*)(ws + off); off += (size_t)EE * DD * F3 * 2;
    __bf16* WprojT = (__bf16*)(ws + off); off += (size_t)EE * DD * DD * 2;

    const size_t na = (size_t)EE * DD * F3;
    const size_t np = (size_t)EE * DD * DD;
    convert_transpose_kernel<<<(unsigned)((na + 255) / 256), 256, 0, stream>>>(
        W_attn, WattnT, DD, F3);
    convert_transpose_kernel<<<(unsigned)((np + 255) / 256), 256, 0, stream>>>(
        W_proj, WprojT, DD, DD);

    router_ln_kernel<<<NN, 256, 0, stream>>>(x, Wr, br, score, xn);

    moe_gemm_kernel<true><<<dim3(NN / 128, F3 / 128), 256, 0, stream>>>(
        xn, score, ln_g, ln_b, WattnT, b_attn, qkvb, DD, F3);

    attn_kernel<<<dim3(BB * HH, SS / 64), 128, 0, stream>>>(qkvb, ctx);

    moe_gemm_kernel<false><<<dim3(NN / 128, DD / 128), 256, 0, stream>>>(
        ctx, score, nullptr, nullptr, WprojT, b_proj, out, DD, DD);
}
